// Corr_13821204759314
// MI455X (gfx1250) — matrix operand order repaired and checked by reading, not compiled, not run on hardware
//
#include <hip/hip_runtime.h>
#include <math.h>

#define BATCH 4
#define CH    256
#define HW    4096   // 64*64 spatial positions = N1 = N2

typedef __attribute__((ext_vector_type(8)))  __bf16          bf16x8;
typedef __attribute__((ext_vector_type(16))) __bf16          bf16x16;
typedef __attribute__((ext_vector_type(8)))  float           f32x8;
typedef __attribute__((ext_vector_type(8)))  unsigned short  u16x8;

__device__ __forceinline__ unsigned short f32_to_bf16_rne(float f) {
    unsigned int u = __float_as_uint(f);
    unsigned int r = 0x7FFFu + ((u >> 16) & 1u);
    return (unsigned short)((u + r) >> 16);
}

// LDS aperture addresses map by truncation to 32 bits (ISA 10.2)
__device__ __forceinline__ unsigned lds_off(const void* p) {
    return (unsigned)(unsigned long long)p;
}

// CDNA5 async copy global -> LDS, per-lane addresses, tracked by ASYNCcnt
__device__ __forceinline__ void async_load_b128(unsigned lds_addr, const void* gaddr) {
    asm volatile("global_load_async_to_lds_b128 %0, %1, off"
                 :: "v"(lds_addr), "v"((unsigned long long)gaddr)
                 : "memory");
}
__device__ __forceinline__ void wait_async0() {
    asm volatile("s_wait_asynccnt 0" ::: "memory");
}

// ---------------------------------------------------------------------------
// K0: channel-L2-normalize [b,c,h,w] f32  ->  bf16 [b, hw, c]  (row-major)
// blockIdx.z selects xp (->P) or xq (->Q). grid (16,4,2) x 256
// ---------------------------------------------------------------------------
__global__ void normalize_kernel(const float* __restrict__ xp,
                                 const float* __restrict__ xq,
                                 unsigned short* __restrict__ P,
                                 unsigned short* __restrict__ Q) {
    const float* x = (blockIdx.z == 0) ? xp : xq;
    unsigned short* o = (blockIdx.z == 0) ? P : Q;
    int b = blockIdx.y;
    int s = blockIdx.x * blockDim.x + threadIdx.x;      // spatial position
    const float* xb = x + (size_t)b * CH * HW + s;
    float ss = 0.0f;
    for (int c = 0; c < CH; ++c) { float v = xb[(size_t)c * HW]; ss += v * v; }
    float inv = 1.0f / fmaxf(sqrtf(ss), 1e-12f);
    unsigned short* ob = o + ((size_t)b * HW + s) * CH;
    for (int c = 0; c < CH; ++c)
        ob[c] = f32_to_bf16_rne(xb[(size_t)c * HW] * inv);
}

// ---------------------------------------------------------------------------
// K1: S[b,i,j] = alpha * sum_k P[b,i,k] * Q[b,j,k]   (bf16 WMMA, f32 acc)
// Block = 256 thr = 8 waves -> 128x64 tile. Operands staged in LDS with
// double-buffered GLOBAL_LOAD_ASYNC_TO_LDS_B128 over K-chunks of 32.
// LDS row stride = 40 u16 (20 dwords) -> conflict-free ds_load_b128 frags.
// grid (64 j-strips, 32 i-blocks, 4 batches).
// ---------------------------------------------------------------------------
#define KC      32                 // K elements per chunk
#define NCHUNK  (CH / KC)          // 8
#define STR     40                 // padded LDS row stride in u16 (80B)

__global__ void __launch_bounds__(256)
gemm_kernel(const unsigned short* __restrict__ P,
            const unsigned short* __restrict__ Q,
            const float* __restrict__ alpha_p,
            float* __restrict__ S) {
    __shared__ unsigned short As[2][128 * STR];   // 2 x 5120 u16
    __shared__ unsigned short Bs[2][64 * STR];    // 2 x  2560 u16

    const float alpha = alpha_p[0];
    const int lane = threadIdx.x & 31;
    const int wave = threadIdx.x >> 5;
    const int half = lane >> 4;          // K-half selector (ISA 7.12.2)
    const int m    = lane & 15;          // A row / B column within tile
    const int b  = blockIdx.z;
    const int iBlock = blockIdx.y * 128;
    const int j0     = blockIdx.x * 64;

    const unsigned short* Pb = P + ((size_t)b * HW + iBlock) * CH;
    const unsigned short* Qb = Q + ((size_t)b * HW + j0) * CH;

    // async-load one K-chunk: each wave stages 16 A rows + 8 B rows.
    // One b128 instr covers 8 rows (lanes/4), 16B sub-column = lane&3.
    const int sub = lane >> 2;           // row within 8-row group
    const int c16 = (lane & 3) * 8;      // u16 column offset within 64B slice
    auto load_chunk = [&](int kk, int buf) {
#pragma unroll
        for (int g = 0; g < 2; ++g) {
            int r = wave * 16 + g * 8 + sub;                 // 0..127
            async_load_b128(lds_off(&As[buf][r * STR + c16]),
                            Pb + (size_t)r * CH + kk + c16);
        }
        int rb = wave * 8 + sub;                             // 0..63
        async_load_b128(lds_off(&Bs[buf][rb * STR + c16]),
                        Qb + (size_t)rb * CH + kk + c16);
    };

    f32x8 acc[4] = {{}, {}, {}, {}};

    load_chunk(0, 0);
    wait_async0();
    __syncthreads();

    for (int c = 0; c < NCHUNK; ++c) {
        const int buf = c & 1;
        if (c + 1 < NCHUNK) load_chunk((c + 1) * KC, buf ^ 1);

        // A fragment: lane<16 holds K = 0..7,16..23 ; lane>=16: 8..15,24..31
        const unsigned short* Arow = &As[buf][(wave * 16 + m) * STR];
        bf16x8 al = __builtin_bit_cast(bf16x8, *(const u16x8*)(Arow + half * 8));
        bf16x8 ah = __builtin_bit_cast(bf16x8, *(const u16x8*)(Arow + 16 + half * 8));
        bf16x16 afrag = __builtin_shufflevector(al, ah,
            0,1,2,3,4,5,6,7,8,9,10,11,12,13,14,15);
#pragma unroll
        for (int t = 0; t < 4; ++t) {
            // B fragment: lane is column; lane<16 holds K = 0..7,16..23 ; lane>=16: 8..15,24..31
            const unsigned short* Brow = &Bs[buf][(t * 16 + m) * STR] + half * 8;
            bf16x8 bl = __builtin_bit_cast(bf16x8, *(const u16x8*)(Brow));
            bf16x8 bh = __builtin_bit_cast(bf16x8, *(const u16x8*)(Brow + 16));
            bf16x16 bfrag = __builtin_shufflevector(bl, bh,
                0,1,2,3,4,5,6,7,8,9,10,11,12,13,14,15);
            acc[t] = __builtin_amdgcn_wmma_f32_16x16x32_bf16(
                false, afrag, false, bfrag, (short)0, acc[t], false, false);
        }

        if (c + 1 < NCHUNK) { wait_async0(); __syncthreads(); }
    }

    // C/D layout: VGPR r -> M = r + 8*half, N = lane&15
    float* Sb = S + (size_t)b * HW * HW;
    const int i0 = iBlock + wave * 16;
#pragma unroll
    for (int r = 0; r < 8; ++r) {
        int row = i0 + half * 8 + r;
        float* srow = Sb + (size_t)row * HW + j0 + m;
        srow[0]  = alpha * acc[0][r];
        srow[16] = alpha * acc[1][r];
        srow[32] = alpha * acc[2][r];
        srow[48] = alpha * acc[3][r];
    }
}

// ---------------------------------------------------------------------------
// K2: per-row max and sum(exp(s-max)); one block per global row (B*HW rows)
// ---------------------------------------------------------------------------
__global__ void rowstats_kernel(const float* __restrict__ S,
                                float* __restrict__ rmax,
                                float* __restrict__ rsum) {
    __shared__ float red[256];
    const int gr = blockIdx.x;
    const float* row = S + (size_t)gr * HW;
    const int t = threadIdx.x;
    float mx = -INFINITY;
    for (int j = t; j < HW; j += 256) mx = fmaxf(mx, row[j]);
    red[t] = mx; __syncthreads();
    for (int s = 128; s > 0; s >>= 1) { if (t < s) red[t] = fmaxf(red[t], red[t + s]); __syncthreads(); }
    const float rowm = red[0]; __syncthreads();
    float sm = 0.0f;
    for (int j = t; j < HW; j += 256) sm += expf(row[j] - rowm);
    red[t] = sm; __syncthreads();
    for (int s = 128; s > 0; s >>= 1) { if (t < s) red[t] += red[t + s]; __syncthreads(); }
    if (t == 0) { rmax[gr] = rowm; rsum[gr] = red[0]; }
}

// ---------------------------------------------------------------------------
// K3: per-column max and sum(exp); thread = column (coalesced row sweeps)
// grid = BATCH*16 blocks x 256 threads
// ---------------------------------------------------------------------------
__global__ void colstats_kernel(const float* __restrict__ S,
                                float* __restrict__ cmax,
                                float* __restrict__ csum) {
    const int b   = blockIdx.x >> 4;
    const int col = ((blockIdx.x & 15) << 8) + threadIdx.x;
    const float* Sb = S + (size_t)b * HW * HW;
    float mx = -INFINITY;
    for (int i = 0; i < HW; ++i) mx = fmaxf(mx, Sb[(size_t)i * HW + col]);
    float sm = 0.0f;
    for (int i = 0; i < HW; ++i) sm += expf(Sb[(size_t)i * HW + col] - mx);
    cmax[b * HW + col] = mx;
    csum[b * HW + col] = sm;
}

// ---------------------------------------------------------------------------
// K4: x = exp(2s - rmax_i - cmax_j) / (rsum_i * csum_j); write xc_o_q in
// place and xc_o_p via LDS 32x33 tile transpose. grid (128,128,4) x (32,8)
// ---------------------------------------------------------------------------
__global__ void finalize_kernel(float* __restrict__ S,       // xc_o_q (in/out)
                                float* __restrict__ XP,      // xc_o_p (out)
                                const float* __restrict__ rmax,
                                const float* __restrict__ rsum,
                                const float* __restrict__ cmax,
                                const float* __restrict__ csum) {
    __shared__ float tile[32][33];
    const int b = blockIdx.z;
    const int row0 = blockIdx.y * 32, col0 = blockIdx.x * 32;
    const int tx = threadIdx.x, ty = threadIdx.y;
    float* Sb = S  + (size_t)b * HW * HW;
    float* Pb = XP + (size_t)b * HW * HW;
    const float* rm = rmax + b * HW; const float* rs = rsum + b * HW;
    const float* cm = cmax + b * HW; const float* cs = csum + b * HW;
#pragma unroll
    for (int r = 0; r < 4; ++r) {
        int row = row0 + ty + r * 8;
        int col = col0 + tx;
        float s = Sb[(size_t)row * HW + col];
        float v = expf(2.0f * s - rm[row] - cm[col]) / (rs[row] * cs[col]);
        Sb[(size_t)row * HW + col] = v;
        tile[ty + r * 8][tx] = v;
    }
    __syncthreads();
#pragma unroll
    for (int r = 0; r < 4; ++r) {
        int prow = col0 + ty + r * 8;    // xc_o_p row = original column
        int pcol = row0 + tx;            // xc_o_p col = original row
        Pb[(size_t)prow * HW + pcol] = tile[tx][ty + r * 8];
    }
}

// ---------------------------------------------------------------------------
// K5: per-row top-3 (sorted descending) with shuffle merge. 1 wave per row.
// out layout [B, 3, HW]. grid = B*HW/8 blocks x 256 threads.
// ---------------------------------------------------------------------------
__device__ __forceinline__ void top3_insert(float v, float& t0, float& t1, float& t2) {
    if (v > t2) {
        if (v > t0)      { t2 = t1; t1 = t0; t0 = v; }
        else if (v > t1) { t2 = t1; t1 = v; }
        else             { t2 = v; }
    }
}

__global__ void topk_kernel(const float* __restrict__ M, float* __restrict__ out) {
    const int gr   = blockIdx.x * 8 + (threadIdx.x >> 5);
    const int lane = threadIdx.x & 31;
    const float* row = M + (size_t)gr * HW;
    float t0 = -INFINITY, t1 = -INFINITY, t2 = -INFINITY;
    for (int j = lane; j < HW; j += 32) top3_insert(row[j], t0, t1, t2);
    for (int off = 16; off > 0; off >>= 1) {
        float b0 = __shfl_down(t0, off);
        float b1 = __shfl_down(t1, off);
        float b2 = __shfl_down(t2, off);
        top3_insert(b0, t0, t1, t2);
        top3_insert(b1, t0, t1, t2);
        top3_insert(b2, t0, t1, t2);
    }
    if (lane == 0) {
        int b = gr >> 12, i = gr & (HW - 1);
        out[((size_t)b * 3 + 0) * HW + i] = t0;
        out[((size_t)b * 3 + 1) * HW + i] = t1;
        out[((size_t)b * 3 + 2) * HW + i] = t2;
    }
}

// ---------------------------------------------------------------------------
extern "C" void kernel_launch(void* const* d_in, const int* in_sizes, int n_in,
                              void* d_out, int out_size, void* d_ws, size_t ws_size,
                              hipStream_t stream) {
    const float* xp    = (const float*)d_in[0];
    const float* xq    = (const float*)d_in[1];
    const float* alpha = (const float*)d_in[2];
    float* out = (float*)d_out;

    // workspace: P(8MB bf16) | Q(8MB bf16) | rmax | rsum | cmax | csum  (~17MB)
    unsigned short* P = (unsigned short*)d_ws;
    unsigned short* Q = P + (size_t)BATCH * HW * CH;
    float* rmax = (float*)((char*)d_ws + (size_t)2 * BATCH * HW * CH * sizeof(unsigned short));
    float* rsum = rmax + (size_t)BATCH * HW;
    float* cmax = rsum + (size_t)BATCH * HW;
    float* csum = cmax + (size_t)BATCH * HW;

    // outputs flat-concat: valp, valq, xc_o_p, xc_o_q
    float* valp = out;
    float* valq = valp + (size_t)BATCH * 3 * HW;
    float* xcp  = valq + (size_t)BATCH * 3 * HW;
    float* xcq  = xcp  + (size_t)BATCH * HW * HW;   // also scratch for S

    normalize_kernel<<<dim3(HW / 256, BATCH, 2), 256, 0, stream>>>(xp, xq, P, Q);
    gemm_kernel<<<dim3(HW / 64, HW / 128, BATCH), 256, 0, stream>>>(P, Q, alpha, xcq);
    rowstats_kernel<<<dim3(BATCH * HW), 256, 0, stream>>>(xcq, rmax, rsum);
    colstats_kernel<<<dim3(BATCH * 16), 256, 0, stream>>>(xcq, cmax, csum);
    finalize_kernel<<<dim3(HW / 32, HW / 32, BATCH), dim3(32, 8), 0, stream>>>(
        xcq, xcp, rmax, rsum, cmax, csum);
    topk_kernel<<<dim3(BATCH * HW / 8), 256, 0, stream>>>(xcq, valp);
    topk_kernel<<<dim3(BATCH * HW / 8), 256, 0, stream>>>(xcp, valq);
}
